// GraphEncoder_73967926771713
// MI455X (gfx1250) — matrix operand order repaired and checked by reading, not compiled, not run on hardware
//
#include <hip/hip_runtime.h>
#include <hip/hip_bf16.h>
#include <math.h>

// ---------------------------------------------------------------------------
// GAT encoder for MI455X (gfx1250, wave32, WMMA).
// GEMMs: v_wmma_f32_16x16x32_f16 with f32 accumulation; A tiles staged with
// GLOBAL_LOAD_ASYNC_TO_LDS_B128 (ASYNCcnt). Attention softmax/scatter uses
// f32 atomics that resolve in the 192MB L2 (h and out buffers both fit).
// ---------------------------------------------------------------------------

typedef __attribute__((ext_vector_type(16))) _Float16 v16h;
typedef __attribute__((ext_vector_type(8)))  float    v8f;

#define TILE_BM 128
#define TILE_BN 64
#define TILE_BK 32

// ---------------------------------------------------------------------------
// Tiled f16 WMMA GEMM: C[M,N] = A[M,K] * B[K,N]  (A,B f16 row-major, C f32)
// 256 threads = 8 waves; wave w computes rows [w*16, w*16+16) of the block
// tile, 4 WMMA 16x16 accumulators across BN=64 columns.
// Requires: K % 32 == 0, N % 64 == 0 (true for all three GEMMs here).
// ---------------------------------------------------------------------------
template <bool FUSE_BIAS_RELU>
__global__ __launch_bounds__(256) void wmma_gemm_f16_kernel(
    const _Float16* __restrict__ A, const _Float16* __restrict__ B,
    const float* __restrict__ bias, float* __restrict__ C,
    int M, int N, int K)
{
    // A tile: 128 rows x 32 halves, row stride 20 dwords (80B): 16B-aligned
    // chunks + conflict-free fragment reads.  B tile: 64 cols x 32 halves
    // (K-major / transposed), col stride 20 dwords.
    __shared__ unsigned int Asu[TILE_BM * 20];
    __shared__ unsigned int Bsu[TILE_BN * 20];

    const int tid  = threadIdx.x;
    const int lane = tid & 31;
    const int wave = tid >> 5;
    const int hlf  = lane >> 4;   // 0: lanes 0-15, 1: lanes 16-31
    const int l16  = lane & 15;

    const int rowBlock = blockIdx.y * TILE_BM;
    const int colBlock = blockIdx.x * TILE_BN;

    // ---- A tile copy plan: 2 threads per row, 32B (2x16B) per thread ----
    const int arow_ld = tid >> 1;       // tile row 0..127
    const int aseg    = tid & 1;        // 16-half segment within the row
    int agr = rowBlock + arow_ld;
    agr = agr < M ? agr : (M - 1);      // clamp: rows >= M are store-guarded
    const _Float16* aptr = A + (size_t)agr * K + aseg * 16;
    const unsigned int lds_a = (unsigned int)(size_t)&Asu[arow_ld * 20 + aseg * 8];

    // ---- B tile copy plan: 1 thread per 16B of a K-row ----
    const int bkr = tid >> 3;           // K row in tile 0..31
    const int bc8 = (tid & 7) * 8;      // first of 8 columns
    unsigned short* Bss = (unsigned short*)Bsu;

    union AccU { v8f v; float f[8]; };
    AccU acc[4];
#pragma unroll
    for (int t = 0; t < 4; ++t)
#pragma unroll
        for (int r = 0; r < 8; ++r) acc[t].f[r] = 0.0f;

    for (int k0 = 0; k0 < K; k0 += TILE_BK) {
        // ---- A tile: async global -> LDS, 2x16B per thread (ASYNCcnt) ----
        const _Float16* ap = aptr + k0;
        asm volatile(
            "global_load_async_to_lds_b128 %0, %2, off\n\t"
            "global_load_async_to_lds_b128 %1, %3, off"
            :: "v"(lds_a), "v"(lds_a + 16), "v"(ap), "v"(ap + 8)
            : "memory");

        // ---- B tile: b128 load + transposed (K-major) LDS scatter ----
        uint4 bv = *(const uint4*)(B + (size_t)(k0 + bkr) * N + colBlock + bc8);
        unsigned int bw[4] = {bv.x, bv.y, bv.z, bv.w};
#pragma unroll
        for (int j = 0; j < 4; ++j) {
            Bss[(bc8 + 2 * j + 0) * 40 + bkr] = (unsigned short)(bw[j] & 0xffffu);
            Bss[(bc8 + 2 * j + 1) * 40 + bkr] = (unsigned short)(bw[j] >> 16);
        }

        asm volatile("s_wait_asynccnt 0" ::: "memory");
        __syncthreads();

        // ---- A fragment: ISA 16-bit A 16x32 layout (8 contiguous dwords) ----
        union { v16h v; unsigned int u[8]; } afrag;
        const int arow = wave * 16 + l16;
#pragma unroll
        for (int v = 0; v < 8; ++v) {
            //  v<4 : k/2 = v + hlf*4    ;  v>=4 : k/2 = 8 + (v-4) + hlf*4
            int ku = (v < 4 ? v : v + 4) + hlf * 4;
            afrag.u[v] = Asu[arow * 20 + ku];
        }
        // ---- 4 column subtiles ----
#pragma unroll
        for (int t = 0; t < 4; ++t) {
            union { v16h v; unsigned int u[8]; } bfrag;
            const int bcol = t * 16 + l16;
#pragma unroll
            for (int v = 0; v < 8; ++v)
                bfrag.u[v] = Bsu[bcol * 20 + (v < 4 ? v : v + 4) + hlf * 4];
            acc[t].v = __builtin_amdgcn_wmma_f32_16x16x32_f16(
                false, afrag.v, false, bfrag.v, (short)0, acc[t].v, false, false);
        }
        __syncthreads();
    }

    // ---- store: C/D layout lanes0-15 -> M=r, lanes16-31 -> M=r+8 ----
    float bval[4];
    if (FUSE_BIAS_RELU) {
#pragma unroll
        for (int t = 0; t < 4; ++t) bval[t] = bias[colBlock + t * 16 + l16];
    }
    const int rbase = rowBlock + wave * 16 + hlf * 8;
    if (rowBlock + TILE_BM <= M) {
        // fast path: full tile, branch-free stores
#pragma unroll
        for (int t = 0; t < 4; ++t) {
            float* cp = C + (size_t)rbase * N + colBlock + t * 16 + l16;
#pragma unroll
            for (int r = 0; r < 8; ++r) {
                float v = acc[t].f[r];
                if (FUSE_BIAS_RELU) {
                    v += bval[t];
                    v = v > 0.0f ? v : 0.0f;
                }
                cp[(size_t)r * N] = v;
            }
        }
    } else {
#pragma unroll
        for (int t = 0; t < 4; ++t) {
            float* cp = C + (size_t)rbase * N + colBlock + t * 16 + l16;
#pragma unroll
            for (int r = 0; r < 8; ++r) {
                if (rbase + r < M) {
                    float v = acc[t].f[r];
                    if (FUSE_BIAS_RELU) {
                        v += bval[t];
                        v = v > 0.0f ? v : 0.0f;
                    }
                    cp[(size_t)r * N] = v;
                }
            }
        }
    }
}

// ---------------------------------------------------------------------------
// Elementwise helpers
// ---------------------------------------------------------------------------
__global__ void f32_to_f16_kernel(const float* __restrict__ in,
                                  _Float16* __restrict__ out, int n) {
    int i = blockIdx.x * blockDim.x + threadIdx.x;
    if (i < n) out[i] = (_Float16)in[i];
}

__global__ void fill_f32_kernel(float* __restrict__ p, float val, int n) {
    int i = blockIdx.x * blockDim.x + threadIdx.x;
    if (i < n) p[i] = val;
}

__global__ void bias_relu_kernel(float* __restrict__ out,
                                 const float* __restrict__ b, int n, int F) {
    int i = blockIdx.x * blockDim.x + threadIdx.x;
    if (i >= n * F) return;
    float v = out[i] + b[i % F];
    out[i] = v > 0.0f ? v : 0.0f;
}

// ---------------------------------------------------------------------------
// Attention: per-(node,head) scores
// ---------------------------------------------------------------------------
__global__ void gat_scores_kernel(const float* __restrict__ h,
                                  const float* __restrict__ a_src,
                                  const float* __restrict__ a_dst,
                                  float* __restrict__ ss, float* __restrict__ sd,
                                  int n, int H, int C) {
    int i = blockIdx.x * blockDim.x + threadIdx.x;
    if (i >= n * H) return;
    int node = i / H, hd = i - node * H;
    const float* hp = h + (size_t)node * H * C + (size_t)hd * C;
    const float* as = a_src + (size_t)hd * C;
    const float* ad = a_dst + (size_t)hd * C;
    float s1 = 0.0f, s2 = 0.0f;
    for (int c = 0; c < C; c += 4) {
        float4 hv = *(const float4*)(hp + c);
        float4 av = *(const float4*)(as + c);
        float4 dv = *(const float4*)(ad + c);
        s1 += hv.x * av.x + hv.y * av.y + hv.z * av.z + hv.w * av.w;
        s2 += hv.x * dv.x + hv.y * dv.y + hv.z * dv.z + hv.w * dv.w;
    }
    ss[i] = s1;
    sd[i] = s2;
}

__device__ __forceinline__ void atomicMaxF(float* addr, float val) {
    if (val >= 0.0f)
        atomicMax((int*)addr, __float_as_int(val));
    else
        atomicMin((unsigned int*)addr, __float_as_uint(val));
}

// edge ids [0,E) come from edge_index, [E, E+n) are self loops
__device__ __forceinline__ void edge_src_dst(const int* ei, int ed, int E,
                                             int& s, int& d) {
    if (ed < E) { s = ei[ed]; d = ei[E + ed]; }
    else        { s = d = ed - E; }
}

__global__ void edge_score_max_kernel(const int* __restrict__ ei,
                                      const float* __restrict__ ss,
                                      const float* __restrict__ sd,
                                      float* __restrict__ ebuf,
                                      float* __restrict__ mbuf,
                                      int E, int n, int H) {
    int i = blockIdx.x * blockDim.x + threadIdx.x;
    if (i >= (E + n) * H) return;
    int ed = i / H, hd = i - ed * H;
    int s, d;
    edge_src_dst(ei, ed, E, s, d);
    float e = ss[s * H + hd] + sd[d * H + hd];
    e = e > 0.0f ? e : e * 0.2f;   // leaky relu, slope 0.2
    ebuf[i] = e;
    atomicMaxF(mbuf + d * H + hd, e);
}

__global__ void edge_exp_sum_kernel(const int* __restrict__ ei,
                                    float* __restrict__ ebuf,
                                    const float* __restrict__ mbuf,
                                    float* __restrict__ den,
                                    int E, int n, int H) {
    int i = blockIdx.x * blockDim.x + threadIdx.x;
    if (i >= (E + n) * H) return;
    int ed = i / H, hd = i - ed * H;
    int s, d;
    edge_src_dst(ei, ed, E, s, d);
    float e = __expf(ebuf[i] - mbuf[d * H + hd]);
    ebuf[i] = e;
    atomicAdd(den + d * H + hd, e);
}

__global__ void edge_scatter_kernel(const int* __restrict__ ei,
                                    const float* __restrict__ ebuf,
                                    const float* __restrict__ den,
                                    const float* __restrict__ h,
                                    float* __restrict__ out,
                                    int E, int n, int H, int C) {
    int i = blockIdx.x * blockDim.x + threadIdx.x;
    if (i >= (E + n) * H) return;
    int ed = i / H, hd = i - ed * H;
    int s, d;
    edge_src_dst(ei, ed, E, s, d);
    float alpha = ebuf[i] / (den[d * H + hd] + 1e-16f);
    const float* hs = h + (size_t)s * H * C + (size_t)hd * C;
    float*       op = out + (size_t)d * H * C + (size_t)hd * C;
#pragma unroll 4
    for (int c = 0; c < C; ++c)
        atomicAdd(op + c, alpha * hs[c]);
}

// ---------------------------------------------------------------------------
// Host-side orchestration
// ---------------------------------------------------------------------------
static inline int cdiv(int a, int b) { return (a + b - 1) / b; }

extern "C" void kernel_launch(void* const* d_in, const int* in_sizes, int n_in,
                              void* d_out, int out_size, void* d_ws, size_t ws_size,
                              hipStream_t stream) {
    const float* x      = (const float*)d_in[0];
    const int*   ei     = (const int*)  d_in[1];
    const float* W1     = (const float*)d_in[2];
    const float* a_src1 = (const float*)d_in[3];
    const float* a_dst1 = (const float*)d_in[4];
    const float* b1     = (const float*)d_in[5];
    const float* W2     = (const float*)d_in[6];
    const float* a_src2 = (const float*)d_in[7];
    const float* a_dst2 = (const float*)d_in[8];
    const float* b2     = (const float*)d_in[9];
    const float* Wfc    = (const float*)d_in[10];
    const float* bfc    = (const float*)d_in[11];

    const int D_IN = 256, HID = 64, HEADS = 10;
    const int N = in_sizes[0] / D_IN;
    const int E = in_sizes[1] / 2;
    const int F1 = HEADS * HID;           // 640
    const int ETOT = E + N;               // with self loops

    // ---- workspace carve ----
    char* ws = (char*)d_ws;
    size_t off = 0;
    auto carve = [&](size_t bytes) -> void* {
        void* p = ws + off;
        off += (bytes + 255) & ~(size_t)255;
        return p;
    };
    _Float16* xh   = (_Float16*)carve((size_t)N * D_IN * 2);
    _Float16* w1h  = (_Float16*)carve((size_t)D_IN * F1 * 2);
    float*    h1   = (float*)   carve((size_t)N * F1 * 4);
    float*    ss1  = (float*)   carve((size_t)N * HEADS * 4);
    float*    sd1  = (float*)   carve((size_t)N * HEADS * 4);
    float*    m1   = (float*)   carve((size_t)N * HEADS * 4);
    float*    den1 = (float*)   carve((size_t)N * HEADS * 4);
    float*    e1   = (float*)   carve((size_t)ETOT * HEADS * 4);
    float*    o1   = (float*)   carve((size_t)N * F1 * 4);
    _Float16* o1h  = (_Float16*)carve((size_t)N * F1 * 2);
    _Float16* w2h  = (_Float16*)carve((size_t)F1 * HID * 2);
    float*    h2   = (float*)   carve((size_t)N * HID * 4);
    float*    ss2  = (float*)   carve((size_t)N * 4);
    float*    sd2  = (float*)   carve((size_t)N * 4);
    float*    m2   = (float*)   carve((size_t)N * 4);
    float*    den2 = (float*)   carve((size_t)N * 4);
    float*    e2   = (float*)   carve((size_t)ETOT * 4);
    float*    o2   = (float*)   carve((size_t)N * HID * 4);
    _Float16* o2h  = (_Float16*)carve((size_t)N * HID * 2);
    _Float16* wfch = (_Float16*)carve((size_t)HID * HID * 2);
    (void)ws_size; (void)n_in; (void)out_size;

    const int TB = 256;
    const float NEG_INF = -INFINITY;

    // ================= Layer 1: GATConv(256 -> 64 x 10 heads) =============
    f32_to_f16_kernel<<<cdiv(N * D_IN, TB), TB, 0, stream>>>(x, xh, N * D_IN);
    f32_to_f16_kernel<<<cdiv(D_IN * F1, TB), TB, 0, stream>>>(W1, w1h, D_IN * F1);

    wmma_gemm_f16_kernel<false><<<dim3(cdiv(F1, TILE_BN), cdiv(N, TILE_BM)), TB, 0, stream>>>(
        xh, w1h, nullptr, h1, N, F1, D_IN);

    fill_f32_kernel<<<cdiv(N * HEADS, TB), TB, 0, stream>>>(m1, NEG_INF, N * HEADS);
    fill_f32_kernel<<<cdiv(N * HEADS, TB), TB, 0, stream>>>(den1, 0.0f, N * HEADS);
    fill_f32_kernel<<<cdiv(N * F1, TB), TB, 0, stream>>>(o1, 0.0f, N * F1);

    gat_scores_kernel<<<cdiv(N * HEADS, TB), TB, 0, stream>>>(
        h1, a_src1, a_dst1, ss1, sd1, N, HEADS, HID);
    edge_score_max_kernel<<<cdiv(ETOT * HEADS, TB), TB, 0, stream>>>(
        ei, ss1, sd1, e1, m1, E, N, HEADS);
    edge_exp_sum_kernel<<<cdiv(ETOT * HEADS, TB), TB, 0, stream>>>(
        ei, e1, m1, den1, E, N, HEADS);
    edge_scatter_kernel<<<cdiv(ETOT * HEADS, TB), TB, 0, stream>>>(
        ei, e1, den1, h1, o1, E, N, HEADS, HID);
    bias_relu_kernel<<<cdiv(N * F1, TB), TB, 0, stream>>>(o1, b1, N, F1);

    // ================= Layer 2: GATConv(640 -> 64, 1 head) =================
    f32_to_f16_kernel<<<cdiv(N * F1, TB), TB, 0, stream>>>(o1, o1h, N * F1);
    f32_to_f16_kernel<<<cdiv(F1 * HID, TB), TB, 0, stream>>>(W2, w2h, F1 * HID);

    wmma_gemm_f16_kernel<false><<<dim3(cdiv(HID, TILE_BN), cdiv(N, TILE_BM)), TB, 0, stream>>>(
        o1h, w2h, nullptr, h2, N, HID, F1);

    fill_f32_kernel<<<cdiv(N, TB), TB, 0, stream>>>(m2, NEG_INF, N);
    fill_f32_kernel<<<cdiv(N, TB), TB, 0, stream>>>(den2, 0.0f, N);
    fill_f32_kernel<<<cdiv(N * HID, TB), TB, 0, stream>>>(o2, 0.0f, N * HID);

    gat_scores_kernel<<<cdiv(N, TB), TB, 0, stream>>>(
        h2, a_src2, a_dst2, ss2, sd2, N, 1, HID);
    edge_score_max_kernel<<<cdiv(ETOT, TB), TB, 0, stream>>>(
        ei, ss2, sd2, e2, m2, E, N, 1);
    edge_exp_sum_kernel<<<cdiv(ETOT, TB), TB, 0, stream>>>(
        ei, e2, m2, den2, E, N, 1);
    edge_scatter_kernel<<<cdiv(ETOT, TB), TB, 0, stream>>>(
        ei, e2, den2, h2, o2, E, N, 1, HID);
    bias_relu_kernel<<<cdiv(N * HID, TB), TB, 0, stream>>>(o2, b2, N, HID);

    // ================= Final Linear(64 -> 64) + ReLU =======================
    f32_to_f16_kernel<<<cdiv(N * HID, TB), TB, 0, stream>>>(o2, o2h, N * HID);
    f32_to_f16_kernel<<<cdiv(HID * HID, TB), TB, 0, stream>>>(Wfc, wfch, HID * HID);

    wmma_gemm_f16_kernel<true><<<dim3(cdiv(HID, TILE_BN), cdiv(N, TILE_BM)), TB, 0, stream>>>(
        o2h, wfch, bfc, (float*)d_out, N, HID, HID);
}
